// OnlineTripletLoss_8796093022285
// MI455X (gfx1250) — compile-verified, matrix operand order checked by reading, not run on hardware
//
#include <hip/hip_runtime.h>
#include <hip/hip_bf16.h>

typedef __attribute__((ext_vector_type(16))) _Float16 v16h;
typedef __attribute__((ext_vector_type(8)))  _Float16 v8h;
typedef __attribute__((ext_vector_type(4)))  _Float16 v4h;
typedef __attribute__((ext_vector_type(8)))  float    v8f;

#define TL_MARGIN 1.0f
#define TL_EPS    1e-12f
#define NEG_INF   (-__builtin_inff())
#define POS_INF   ( __builtin_inff())

// ---------------------------------------------------------------------------
// Kernel 1: row L2-normalize, convert to f16, compute sq[i] = sum(f16(e)^2)
// One wave32 per row. K must be a multiple of 128 (768 ok).
// ---------------------------------------------------------------------------
__global__ void tl_normalize_kernel(const float* __restrict__ x,
                                    _Float16* __restrict__ eh,
                                    float* __restrict__ sq,
                                    int N, int K) {
    int row  = (blockIdx.x * blockDim.x + threadIdx.x) >> 5;
    int lane = threadIdx.x & 31;
    if (row >= N) return;

    const float* xr = x + (size_t)row * K;

    float ss = 0.0f;
    for (int k = lane * 4; k < K; k += 128) {
        float4 v = *reinterpret_cast<const float4*>(xr + k);
        ss += v.x * v.x + v.y * v.y + v.z * v.z + v.w * v.w;
    }
    #pragma unroll
    for (int m = 16; m >= 1; m >>= 1) ss += __shfl_xor(ss, m, 32);

    float inv = 1.0f / fmaxf(sqrtf(ss), TL_EPS);

    _Float16* er = eh + (size_t)row * K;
    float ss2 = 0.0f;
    for (int k = lane * 4; k < K; k += 128) {
        float4 v = *reinterpret_cast<const float4*>(xr + k);
        v4h h;
        h[0] = (_Float16)(v.x * inv);
        h[1] = (_Float16)(v.y * inv);
        h[2] = (_Float16)(v.z * inv);
        h[3] = (_Float16)(v.w * inv);
        *reinterpret_cast<v4h*>(er + k) = h;
        float f0 = (float)h[0], f1 = (float)h[1], f2 = (float)h[2], f3 = (float)h[3];
        ss2 += f0 * f0 + f1 * f1 + f2 * f2 + f3 * f3;
    }
    #pragma unroll
    for (int m = 16; m >= 1; m >>= 1) ss2 += __shfl_xor(ss2, m, 32);

    if (lane == 0) sq[row] = ss2;
}

// ---------------------------------------------------------------------------
// Fragment loader, uniform byte base + 32-bit byte offset (no address
// scaling -> saddr+voffset-friendly). 16x32 f16 A/B WMMA layout: lane holds
// row/col (lane&15); half = lane>>4 selects K sub-chunks [k+8h, +8) and
// [k+16+8h, +8). Both chunks 16-byte aligned -> two global_load_b128.
// ---------------------------------------------------------------------------
__device__ __forceinline__ v16h tl_frag(const uint8_t* __restrict__ base,
                                        uint32_t boff) {
    v8h lo = *reinterpret_cast<const v8h*>(base + boff);
    v8h hi = *reinterpret_cast<const v8h*>(base + boff + 32);
    v16h r;
    #pragma unroll
    for (int t = 0; t < 8; ++t) { r[t] = lo[t]; r[t + 8] = hi[t]; }
    return r;
}

// ---------------------------------------------------------------------------
// Kernel 2: fused WMMA Gram-tile GEMM + distance + batch-hard mining.
// Block = 256 threads = 8 waves; one block per 16-anchor row tile.
// Each wave computes a 16x64 C tile (4 column tiles at once): A fragment
// loaded once per k-step, reused for 4 WMMAs -> 2.5 b128 loads per WMMA.
// All 10 b128 loads of a k-step are issued before the first WMMA so the
// matrix pipe runs against partial load waits. Only acc[4], offsets and
// pmax/nmin are live across the k-loop (no spills); epilogue metadata is
// reloaded per group. The NxN distance matrix never touches memory.
// ---------------------------------------------------------------------------
__global__ void tl_mine_kernel(const _Float16* __restrict__ eh,
                               const float* __restrict__ sq,
                               const int* __restrict__ labels,
                               float* __restrict__ d_ap,
                               float* __restrict__ d_an,
                               int N, int K) {
    const int rbase   = blockIdx.x * 16;
    const int wave    = threadIdx.x >> 5;
    const int lane    = threadIdx.x & 31;
    const int col_in  = lane & 15;
    const int halfid  = lane >> 4;
    const int halfoff = halfid * 8;
    const int ntiles  = N >> 4;   // 256 for N=4096

    const uint8_t* ehb = reinterpret_cast<const uint8_t*>(eh);
    const uint32_t rowbytes = (uint32_t)K * 2u;

    // A-row byte offset for this lane (fits in 32 bits: N*K*2 = 6.3 MB).
    const uint32_t offA = (uint32_t)(rbase + col_in) * rowbytes + (uint32_t)halfoff * 2u;

    float pmax[8], nmin[8];
    #pragma unroll
    for (int v = 0; v < 8; ++v) { pmax[v] = NEG_INF; nmin[v] = POS_INF; }

    // Wave w owns column-tile groups {w*4 .. w*4+3}, stepping by 32 tiles.
    for (int ct0 = wave * 4; ct0 < ntiles; ct0 += 32) {
        uint32_t offB[4];
        #pragma unroll
        for (int c = 0; c < 4; ++c)
            offB[c] = (uint32_t)((ct0 + c) * 16 + col_in) * rowbytes
                    + (uint32_t)halfoff * 2u;

        v8f acc[4] = {};
        #pragma unroll 2
        for (int kb = 0; kb < (int)rowbytes; kb += 64) {      // 32 f16 per step
            // Issue all 10 b128 loads up front, then the 4 WMMAs: the
            // scheduler can overlap WMMA execution with outstanding loads.
            v16h a  = tl_frag(ehb, offA + kb);
            v16h b0 = tl_frag(ehb, offB[0] + kb);
            v16h b1 = tl_frag(ehb, offB[1] + kb);
            v16h b2 = tl_frag(ehb, offB[2] + kb);
            v16h b3 = tl_frag(ehb, offB[3] + kb);
            acc[0] = __builtin_amdgcn_wmma_f32_16x16x32_f16(
                false, a, false, b0, (short)0, acc[0], false, false);
            acc[1] = __builtin_amdgcn_wmma_f32_16x16x32_f16(
                false, a, false, b1, (short)0, acc[1], false, false);
            acc[2] = __builtin_amdgcn_wmma_f32_16x16x32_f16(
                false, a, false, b2, (short)0, acc[2], false, false);
            acc[3] = __builtin_amdgcn_wmma_f32_16x16x32_f16(
                false, a, false, b3, (short)0, acc[3], false, false);
        }

        // Epilogue: reload per-row / per-col metadata (L2-resident, cheap).
        float sqi[8];
        int   li[8];
        #pragma unroll
        for (int v = 0; v < 8; ++v) {
            int i  = rbase + v + 8 * halfid;
            sqi[v] = sq[i];
            li[v]  = labels[i];
        }

        #pragma unroll
        for (int c = 0; c < 4; ++c) {
            const int   j   = (ct0 + c) * 16 + col_in;
            const float sqj = sq[j];
            const int   lj  = labels[j];
            #pragma unroll
            for (int v = 0; v < 8; ++v) {
                int   i = rbase + v + 8 * halfid;
                float d = sqi[v] - 2.0f * acc[c][v] + sqj;
                d = sqrtf(fmaxf(d, 0.0f) + TL_EPS);
                bool same = (li[v] == lj);
                if (same) {
                    if (i != j) pmax[v] = fmaxf(pmax[v], d);
                } else {
                    nmin[v] = fminf(nmin[v], d);
                }
            }
        }
    }

    // Cross-lane reduce within each 16-lane half (rows differ between halves,
    // xor masks < 16 never cross the half boundary).
    #pragma unroll
    for (int m = 8; m >= 1; m >>= 1) {
        #pragma unroll
        for (int v = 0; v < 8; ++v) {
            pmax[v] = fmaxf(pmax[v], __shfl_xor(pmax[v], m, 32));
            nmin[v] = fminf(nmin[v], __shfl_xor(nmin[v], m, 32));
        }
    }

    __shared__ float s_pmax[8][16];
    __shared__ float s_nmin[8][16];
    if (col_in == 0) {
        #pragma unroll
        for (int v = 0; v < 8; ++v) {
            s_pmax[wave][v + 8 * halfid] = pmax[v];
            s_nmin[wave][v + 8 * halfid] = nmin[v];
        }
    }
    __syncthreads();

    if (threadIdx.x < 16) {
        float pm = NEG_INF, nm = POS_INF;
        #pragma unroll
        for (int w = 0; w < 8; ++w) {
            pm = fmaxf(pm, s_pmax[w][threadIdx.x]);
            nm = fminf(nm, s_nmin[w][threadIdx.x]);
        }
        d_ap[rbase + threadIdx.x] = pm;
        d_an[rbase + threadIdx.x] = nm;
    }
}

// ---------------------------------------------------------------------------
// Kernel 3: final scalar reduction (single block).
// ---------------------------------------------------------------------------
__global__ void tl_loss_kernel(const float* __restrict__ d_ap,
                               const float* __restrict__ d_an,
                               float* __restrict__ out, int N) {
    float sum = 0.0f, cnt = 0.0f;
    for (int i = threadIdx.x; i < N; i += blockDim.x) {
        float pm = d_ap[i];
        if (pm > -1e30f) {  // anchor has at least one positive
            cnt += 1.0f;
            sum += fmaxf(pm - d_an[i] + TL_MARGIN, 0.0f);
        }
    }
    #pragma unroll
    for (int m = 16; m >= 1; m >>= 1) {
        sum += __shfl_xor(sum, m, 32);
        cnt += __shfl_xor(cnt, m, 32);
    }
    __shared__ float s_sum[8], s_cnt[8];
    int wave = threadIdx.x >> 5, lane = threadIdx.x & 31;
    if (lane == 0) { s_sum[wave] = sum; s_cnt[wave] = cnt; }
    __syncthreads();
    if (threadIdx.x == 0) {
        float ts = 0.0f, tc = 0.0f;
        for (int w = 0; w < 8; ++w) { ts += s_sum[w]; tc += s_cnt[w]; }
        out[0] = ts / fmaxf(tc, 1.0f);
    }
}

// ---------------------------------------------------------------------------
// Launcher
// ---------------------------------------------------------------------------
extern "C" void kernel_launch(void* const* d_in, const int* in_sizes, int n_in,
                              void* d_out, int out_size, void* d_ws, size_t ws_size,
                              hipStream_t stream) {
    const float* embeddings = (const float*)d_in[0];
    const int*   labels     = (const int*)d_in[1];
    float*       out        = (float*)d_out;

    const int N = in_sizes[1];            // 4096
    const int K = in_sizes[0] / N;        // 768

    // Workspace layout
    char*     ws = (char*)d_ws;
    _Float16* eh = (_Float16*)ws;                      // N*K f16 (16B aligned)
    size_t    eh_bytes = ((size_t)N * K * sizeof(_Float16) + 255) & ~(size_t)255;
    float*    sq   = (float*)(ws + eh_bytes);          // N floats
    float*    d_ap = sq + N;                           // N floats
    float*    d_an = d_ap + N;                         // N floats

    // 1) normalize + f16 convert + sq  (8 rows per 256-thread block)
    {
        int blocks = (N + 7) / 8;
        tl_normalize_kernel<<<blocks, 256, 0, stream>>>(embeddings, eh, sq, N, K);
    }

    // 2) fused WMMA GEMM + distances + batch-hard mining (1 block per 16 rows)
    {
        int blocks = N / 16;
        tl_mine_kernel<<<blocks, 256, 0, stream>>>(eh, sq, labels, d_ap, d_an, N, K);
    }

    // 3) scalar loss
    tl_loss_kernel<<<1, 256, 0, stream>>>(d_ap, d_an, out, N);
}
